// MultiScaleAttention_65171833749625
// MI455X (gfx1250) — matrix operand order repaired and checked by reading, not compiled, not run on hardware
//
#include <hip/hip_runtime.h>
#include <hip/hip_bf16.h>
#include <stdint.h>

typedef _Float16 h8   __attribute__((ext_vector_type(8)));
typedef _Float16 v16h __attribute__((ext_vector_type(16)));
typedef float    v8f  __attribute__((ext_vector_type(8)));
typedef unsigned int u32x4 __attribute__((ext_vector_type(4)));
typedef int          i32x4 __attribute__((ext_vector_type(4)));
typedef int          i32x8 __attribute__((ext_vector_type(8)));

#define BH    32
#define Dh    64
#define LFULL 2048

#if defined(__has_builtin)
#if __has_builtin(__builtin_amdgcn_tensor_load_to_lds)
#define MSA_USE_TDM 1
#endif
#endif
#ifndef MSA_USE_TDM
#define MSA_USE_TDM 0
#endif

union U16h { v16h v; h8 h[2]; };

// 16-lane butterfly max all-reduce: four fused v_max_num_f32_dpp instructions.
// VOP2+DPP computes dst = max(dpp(src0), src1) in a single VALU op, avoiding the
// separate v_mov_b32_dpp + canonicalize the generic builtin lowering produces.
__device__ __forceinline__ float msa_rowmax16(float v) {
    float a;
    asm("v_max_num_f32_dpp %0, %1, %1 quad_perm:[1,0,3,2] row_mask:0xf bank_mask:0xf bound_ctrl:1"
        : "=v"(a) : "v"(v));
    asm("v_max_num_f32_dpp %0, %1, %1 quad_perm:[2,3,0,1] row_mask:0xf bank_mask:0xf bound_ctrl:1"
        : "=v"(v) : "v"(a));
    asm("v_max_num_f32_dpp %0, %1, %1 row_half_mirror row_mask:0xf bank_mask:0xf bound_ctrl:1"
        : "=v"(a) : "v"(v));
    asm("v_max_num_f32_dpp %0, %1, %1 row_mirror row_mask:0xf bank_mask:0xf bound_ctrl:1"
        : "=v"(v) : "v"(a));
    return v;
}

// ---------------- f32 -> f16 mean-pool (scale s), optional output scaling ----------------
__global__ void msa_pool_f16(const float* __restrict__ src, _Float16* __restrict__ dst,
                             int Ls, int s, float oscale) {
    int idx = blockIdx.x * blockDim.x + threadIdx.x;
    int total = BH * Ls * Dh;
    if (idx >= total) return;
    int d  = idx & (Dh - 1);
    int i  = (idx >> 6) % Ls;
    int bh = idx / (Ls * Dh);
    const float* p = src + (size_t)bh * LFULL * Dh + (size_t)(i * s) * Dh + d;
    float acc = 0.f;
    for (int t = 0; t < s; ++t) acc += p[t * Dh];
    dst[idx] = (_Float16)(acc * oscale / (float)s);
}

// ---------------- flash attention, f16 WMMA, f32 accum ----------------
// Q is pre-scaled by (1/sqrt(D))*log2(e), so scores are already in the exp2 domain.
// K-dim permutation for the P*V GEMM: score col (tile t, n) lives at physical
// p = (t>>1)*32 + 2n + (t&1); V rows are stored at the same p. WMMA only needs
// A and B to agree on k-ordering, so any fixed permutation is valid.
__launch_bounds__(128)
__global__ void msa_flash_wmma(const _Float16* __restrict__ Q,
                               const _Float16* __restrict__ K,
                               const _Float16* __restrict__ V,
                               float* __restrict__ O,
                               int Ls) {
    __shared__ alignas(32) _Float16 Ksh[64 * 64];      // K block, row-major [krow][d]
    __shared__ alignas(32) _Float16 Vsh[80 * 64];      // [n][p]: rows 0..63 = V^T (perm k),
                                                       // rows 64..79 = ones (row-sum tile)
    __shared__ alignas(32) _Float16 Psh[4][16 * 64];   // per-wave P transpose tile

    const int tid  = threadIdx.x;
    const int wave = tid >> 5;
    const int lane = tid & 31;
    const int half = lane >> 4;
    const int l16  = lane & 15;

    const int bh    = blockIdx.y;
    const int qbase = blockIdx.x * 64 + wave * 16;

    const size_t headoff = (size_t)bh * Ls * Dh;
    const _Float16* Qh = Q + headoff;
    const _Float16* Kh = K + headoff;
    const _Float16* Vh = V + headoff;

    // ones rows for the row-sum tile (written once; staging never touches rows 64..79)
    {
        h8 one;
        #pragma unroll
        for (int m = 0; m < 8; ++m) one[m] = (_Float16)1.0f;
        *(h8*)(Vsh + 64 * 64 + tid * 8) = one;
    }

    // Q A-fragments (16x32 f16 each): lane row m = l16, elem e -> d = c*32+half*16+e
    U16h aq0, aq1;
    {
        const _Float16* qrow = Qh + (size_t)(qbase + l16) * Dh;
        aq0.h[0] = *(const h8*)(qrow + 0  + half * 16);
        aq0.h[1] = *(const h8*)(qrow + 8  + half * 16);
        aq1.h[0] = *(const h8*)(qrow + 32 + half * 16);
        aq1.h[1] = *(const h8*)(qrow + 40 + half * 16);
    }

    v8f oacc[5];
    #pragma unroll
    for (int t = 0; t < 5; ++t) oacc[t] = (v8f){};
    float mstat[8];
    #pragma unroll
    for (int r = 0; r < 8; ++r) mstat[r] = -1e30f;

    const int nIter = Ls >> 6;
    for (int j = 0; j < nIter; ++j) {
        // ---- stage 64x64 K block: Tensor Data Mover (2D tile DMA) if available ----
#if MSA_USE_TDM
        if (wave == 0) {
            unsigned long long ga =
                (unsigned long long)(uintptr_t)(Kh + (size_t)(j * 64) * Dh);
            unsigned int lds = (unsigned int)(uintptr_t)(&Ksh[0]);
            u32x4 g0;
            g0[0] = 1u;                                    // count=1, load, no gather
            g0[1] = lds;                                   // lds_addr
            g0[2] = (unsigned int)ga;                      // global_addr[31:0]
            g0[3] = (unsigned int)((ga >> 32) & 0x01FFFFFFu) | (2u << 30); // [56:32]|type=2
            i32x8 g1;
            g1[0] = 1 << 16;        // workgroup_mask=0, data_size=1 (2B)
            g1[1] = 64 << 16;       // tensor_dim0 = 64 (bits 79:48 low half)
            g1[2] = 64 << 16;       // tensor_dim0 hi=0, tensor_dim1 = 64
            g1[3] = 64 << 16;       // tensor_dim1 hi=0, tile_dim0 = 64
            g1[4] = 64;             // tile_dim1 = 64, tile_dim2 = 0
            g1[5] = 64;             // tensor_dim0_stride = 64
            g1[6] = 0;              // stride0 hi, stride1 lo
            g1[7] = 0;              // stride1 hi
            i32x4 gz = (i32x4){0, 0, 0, 0};
#if __clang_major__ >= 23
            __builtin_amdgcn_tensor_load_to_lds(g0, g1, gz, gz,
                                                (i32x8){0, 0, 0, 0, 0, 0, 0, 0}, 0);
#else
            __builtin_amdgcn_tensor_load_to_lds(g0, g1, gz, gz, 0);
#endif
        }
#else
        for (int t = tid; t < 512; t += 128) {
            int krow = t >> 3;
            int col8 = (t & 7) << 3;
            *(h8*)(Ksh + krow * 64 + col8) =
                *(const h8*)(Kh + (size_t)(j * 64 + krow) * Dh + col8);
        }
#endif
        // ---- stage 64x64 V block transposed with k-permutation ----
        for (int t = tid; t < 512; t += 128) {
            int krow = t >> 3;
            int col8 = (t & 7) << 3;
            int tt = krow >> 4, nn = krow & 15;
            int p = ((tt >> 1) << 5) | (nn << 1) | (tt & 1);
            h8 vv = *(const h8*)(Vh + (size_t)(j * 64 + krow) * Dh + col8);
            #pragma unroll
            for (int m = 0; m < 8; ++m)
                Vsh[(col8 + m) * 64 + p] = vv[m];
        }
        if (j + 1 < nIter) {  // global_prefetch_b8 of next block
            const char* pk = (const char*)(Kh + (size_t)((j + 1) * 64) * Dh);
            const char* pv = (const char*)(Vh + (size_t)((j + 1) * 64) * Dh);
            __builtin_prefetch(pk + tid * 64, 0, 1);
            __builtin_prefetch(pv + tid * 64, 0, 1);
        }
#if MSA_USE_TDM
        if (wave == 0) __builtin_amdgcn_s_wait_tensorcnt(0);
#endif
        __syncthreads();

        // ---- S = Q * K^T : four 16x16 tiles, chained over D in K=32 chunks ----
        v8f sc[4];
        #pragma unroll
        for (int t = 0; t < 4; ++t) {
            const _Float16* kb = Ksh + (t * 16 + l16) * 64;
            U16h b0, b1;
            b0.v = *(const v16h*)(kb + 0  + half * 16);
            b1.v = *(const v16h*)(kb + 32 + half * 16);
            v8f acc = (v8f){};
            acc = __builtin_amdgcn_wmma_f32_16x16x32_f16(false, aq0.v, false, b0.v,
                                                         (short)0, acc, false, false);
            acc = __builtin_amdgcn_wmma_f32_16x16x32_f16(false, aq1.v, false, b1.v,
                                                         (short)0, acc, false, false);
            sc[t] = acc;
        }

        // ---- online softmax: running max only (fused DPP16 butterfly);
        //      the row-sum comes from the ones tile on the matrix pipe ----
        _Float16* pw = Psh[wave];
        #pragma unroll
        for (int r = 0; r < 8; ++r) {
            float x0 = sc[0][r], x1 = sc[1][r], x2 = sc[2][r], x3 = sc[3][r];
            float mv = fmaxf(fmaxf(x0, x1), fmaxf(x2, x3));
            mv = msa_rowmax16(mv);
            float mnew  = fmaxf(mstat[r], mv);
            float alpha = exp2f(mstat[r] - mnew);
            mstat[r] = mnew;
            oacc[0][r] *= alpha; oacc[1][r] *= alpha; oacc[2][r] *= alpha;
            oacc[3][r] *= alpha; oacc[4][r] *= alpha;
            union { _Float16 f[2]; unsigned int u; } pka, pkb;
            pka.f[0] = (_Float16)exp2f(x0 - mnew);   // tile0 -> p = 2n
            pka.f[1] = (_Float16)exp2f(x1 - mnew);   // tile1 -> p = 2n+1
            pkb.f[0] = (_Float16)exp2f(x2 - mnew);   // tile2 -> p = 32+2n
            pkb.f[1] = (_Float16)exp2f(x3 - mnew);   // tile3 -> p = 32+2n+1
            int m = r + half * 8;
            *(unsigned int*)(pw + m * 64 + (l16 << 1))      = pka.u;
            *(unsigned int*)(pw + m * 64 + 32 + (l16 << 1)) = pkb.u;
        }

        // A-fragments of P (16x64, k-permuted); wave-internal LDS ordering
        U16h ap0, ap1;
        ap0.h[0] = *(const h8*)(pw + l16 * 64 + 0  + half * 16);
        ap0.h[1] = *(const h8*)(pw + l16 * 64 + 8  + half * 16);
        ap1.h[0] = *(const h8*)(pw + l16 * 64 + 32 + half * 16);
        ap1.h[1] = *(const h8*)(pw + l16 * 64 + 40 + half * 16);

        // ---- O += P(16x64) * V(64x16) per tile; tile 4 accumulates row-sums ----
        #pragma unroll
        for (int dt = 0; dt < 5; ++dt) {
            const _Float16* vb = Vsh + (dt * 16 + l16) * 64;
            U16h b0, b1;
            b0.v = *(const v16h*)(vb + 0  + half * 16);
            b1.v = *(const v16h*)(vb + 32 + half * 16);
            oacc[dt] = __builtin_amdgcn_wmma_f32_16x16x32_f16(false, ap0.v, false, b0.v,
                                                              (short)0, oacc[dt],
                                                              false, false);
            oacc[dt] = __builtin_amdgcn_wmma_f32_16x16x32_f16(false, ap1.v, false, b1.v,
                                                              (short)0, oacc[dt],
                                                              false, false);
        }
        __syncthreads();
    }

    // ---- epilogue: normalize by the matrix-accumulated row-sum and store ----
    float* Oh = O + headoff;
    #pragma unroll
    for (int r = 0; r < 8; ++r) {
        int m = r + half * 8;
        float inv_l = 1.0f / oacc[4][r];
        float* orow = Oh + (size_t)(qbase + m) * Dh;
        orow[0  + l16] = oacc[0][r] * inv_l;
        orow[16 + l16] = oacc[1][r] * inv_l;
        orow[32 + l16] = oacc[2][r] * inv_l;
        orow[48 + l16] = oacc[3][r] * inv_l;
    }
}

// ---------------- weighted blend with upsample ----------------
__global__ void msa_combine(float* __restrict__ out,
                            const float* __restrict__ o2,
                            const float* __restrict__ o4,
                            const float* __restrict__ sw) {
    int idx = blockIdx.x * blockDim.x + threadIdx.x;
    const int total = BH * LFULL * Dh;
    if (idx >= total) return;
    float s0 = sw[0], s1 = sw[1], s2 = sw[2];
    float mx = fmaxf(s0, fmaxf(s1, s2));
    float e0 = __expf(s0 - mx), e1 = __expf(s1 - mx), e2 = __expf(s2 - mx);
    float inv = 1.0f / (e0 + e1 + e2);
    float w0 = e0 * inv, w1 = e1 * inv, w2 = e2 * inv;
    int d   = idx & 63;
    int pos = (idx >> 6) & (LFULL - 1);
    int bh  = idx >> 17;
    int i2 = (bh * (LFULL / 2) + (pos >> 1)) * Dh + d;
    int i4 = (bh * (LFULL / 4) + (pos >> 2)) * Dh + d;
    out[idx] = w0 * out[idx] + w1 * o2[i2] + w2 * o4[i4];
}

extern "C" void kernel_launch(void* const* d_in, const int* in_sizes, int n_in,
                              void* d_out, int out_size, void* d_ws, size_t ws_size,
                              hipStream_t stream) {
    const float* q  = (const float*)d_in[0];
    const float* k  = (const float*)d_in[1];
    const float* v  = (const float*)d_in[2];
    const float* sw = (const float*)d_in[3];
    float* out = (float*)d_out;

    const int L1 = 2048, L2 = 1024, L4 = 512;

    char* ws = (char*)d_ws;
    size_t off = 0;
    auto take = [&](size_t bytes) -> char* {
        char* p = ws + off;
        off += (bytes + 255) & ~(size_t)255;
        return p;
    };
    _Float16* Q1 = (_Float16*)take((size_t)BH * L1 * Dh * 2);
    _Float16* K1 = (_Float16*)take((size_t)BH * L1 * Dh * 2);
    _Float16* V1 = (_Float16*)take((size_t)BH * L1 * Dh * 2);
    _Float16* Q2 = (_Float16*)take((size_t)BH * L2 * Dh * 2);
    _Float16* K2 = (_Float16*)take((size_t)BH * L2 * Dh * 2);
    _Float16* V2 = (_Float16*)take((size_t)BH * L2 * Dh * 2);
    _Float16* Q4 = (_Float16*)take((size_t)BH * L4 * Dh * 2);
    _Float16* K4 = (_Float16*)take((size_t)BH * L4 * Dh * 2);
    _Float16* V4 = (_Float16*)take((size_t)BH * L4 * Dh * 2);
    float* O2 = (float*)take((size_t)BH * L2 * Dh * 4);
    float* O4 = (float*)take((size_t)BH * L4 * Dh * 4);

    // Fold 1/sqrt(D) * log2(e) into Q so scores land directly in the exp2 domain.
    const float qscale = 0.125f * 1.4426950408889634f;

    {
        int t1 = BH * L1 * Dh, t2 = BH * L2 * Dh, t4 = BH * L4 * Dh;
        msa_pool_f16<<<(t1 + 255) / 256, 256, 0, stream>>>(q, Q1, L1, 1, qscale);
        msa_pool_f16<<<(t1 + 255) / 256, 256, 0, stream>>>(k, K1, L1, 1, 1.0f);
        msa_pool_f16<<<(t1 + 255) / 256, 256, 0, stream>>>(v, V1, L1, 1, 1.0f);
        msa_pool_f16<<<(t2 + 255) / 256, 256, 0, stream>>>(q, Q2, L2, 2, qscale);
        msa_pool_f16<<<(t2 + 255) / 256, 256, 0, stream>>>(k, K2, L2, 2, 1.0f);
        msa_pool_f16<<<(t2 + 255) / 256, 256, 0, stream>>>(v, V2, L2, 2, 1.0f);
        msa_pool_f16<<<(t4 + 255) / 256, 256, 0, stream>>>(q, Q4, L4, 4, qscale);
        msa_pool_f16<<<(t4 + 255) / 256, 256, 0, stream>>>(k, K4, L4, 4, 1.0f);
        msa_pool_f16<<<(t4 + 255) / 256, 256, 0, stream>>>(v, V4, L4, 4, 1.0f);
    }

    // flash attention per scale (scale 1 writes d_out directly)
    msa_flash_wmma<<<dim3(L1 / 64, BH), 128, 0, stream>>>(Q1, K1, V1, out, L1);
    msa_flash_wmma<<<dim3(L2 / 64, BH), 128, 0, stream>>>(Q2, K2, V2, O2, L2);
    msa_flash_wmma<<<dim3(L4 / 64, BH), 128, 0, stream>>>(Q4, K4, V4, O4, L4);

    // weighted blend with nearest-neighbor upsample
    {
        int tot = BH * LFULL * Dh;
        msa_combine<<<(tot + 255) / 256, 256, 0, stream>>>(out, O2, O4, sw);
    }
}
